// MultiHeadSelfAttention_15066745274460
// MI455X (gfx1250) — matrix operand order repaired and checked by reading, not compiled, not run on hardware
//
#include <hip/hip_runtime.h>
#include <hip/hip_bf16.h>
#include <math.h>
#include <stdint.h>

// ---------------------------------------------------------------------------
// MI455X (gfx1250) multi-head self-attention, bf16 WMMA pipeline.
//   f32->bf16 convert
//   QKV GEMM (TDM-staged LDS tiles, double buffered) + bias + per-head LN
//   exact-softmax attention holding 32 full score rows in 320KB LDS
//   output projection GEMM (TDM-staged)
// All matmuls: v_wmma_f32_16x16x32_bf16, fp32 accumulate.
// Tile staging: tensor_load_to_lds (TDM) with hardware LDS row padding,
// tracked via TENSORcnt (s_wait_tensorcnt).
// ---------------------------------------------------------------------------

typedef __bf16 bf16_t;
typedef __attribute__((ext_vector_type(16))) __bf16        v16bf;
typedef __attribute__((ext_vector_type(8)))  __bf16        v8bf;
typedef __attribute__((ext_vector_type(8)))  float         v8f;
typedef __attribute__((ext_vector_type(4)))  unsigned int  v4u;
typedef __attribute__((ext_vector_type(8)))  int           v8i;
typedef __attribute__((ext_vector_type(4)))  int           v4i;

#define BB 2
#define SS 2048
#define DD 1024
#define HH 16
#define HD 64
#define TOK (BB * SS)   // 4096 tokens
#define N3  (3 * DD)    // 3072

// LDS tile pitch: TDM pads each 128B row with 16B -> 144B = 72 bf16 elements.
// 144B = 36 banks; gcd(36,64)=4 so 16 consecutive rows hit distinct bank sets.
#define TPITCH 72

static __device__ __forceinline__ v8f wmma_bf(v16bf a, v16bf b, v8f c) {
  // D = A(16x32 bf16) * B(32x16 bf16) + C(16x16 f32)
  return __builtin_amdgcn_wmma_f32_16x16x32_bf16(false, a, false, b, (short)0, c,
                                                 false, false);
}

static __device__ __forceinline__ v16bf combine16(v8bf lo, v8bf hi) {
  v16bf r;
#pragma unroll
  for (int i = 0; i < 8; ++i) { r[i] = lo[i]; r[i + 8] = hi[i]; }
  return r;
}

// A fragment 16x32 (MxK), source row-major [*, ld] bf16 (global or LDS).
// Lane layout (wave32): m = lane&15; elems 0..7 -> k = kbase + (lane&16?8:0)+e,
// elems 8..15 -> +16.  Two contiguous 16B loads per lane.
static __device__ __forceinline__ v16bf load_frag_a(const bf16_t* p, int ld,
                                                    int rbase, int kbase, int lane) {
  const bf16_t* rp = p + (size_t)(rbase + (lane & 15)) * ld + kbase +
                     ((lane & 16) ? 8 : 0);
  return combine16(*(const v8bf*)rp, *(const v8bf*)(rp + 16));
}

// B fragment 32x16 (KxN) where the source stores N as rows over K (K-major
// rows): src[n][k].  Lane: n = nbase + (lane&15);
// elems 0..7 -> k = kbase + (lane&16?8:0) + e, elems 8..15 -> +16.  Two 16B loads.
static __device__ __forceinline__ v16bf load_frag_b(const bf16_t* p, int ld,
                                                    int nbase, int kbase, int lane) {
  const bf16_t* rp = p + (size_t)(nbase + (lane & 15)) * ld + kbase +
                     ((lane & 16) ? 8 : 0);
  return combine16(*(const v8bf*)rp, *(const v8bf*)(rp + 16));
}

// A fragment from fp32 LDS (probability matrix P), converted to bf16.
static __device__ __forceinline__ v16bf load_frag_a_f32(const float* p, int ld,
                                                        int rbase, int kbase, int lane) {
  const float* rp = p + (size_t)(rbase + (lane & 15)) * ld + kbase +
                    ((lane & 16) ? 8 : 0);
  v16bf r;
#pragma unroll
  for (int i = 0; i < 8; ++i) {
    r[i]     = (bf16_t)rp[i];
    r[i + 8] = (bf16_t)rp[i + 16];
  }
  return r;
}

// ---------------------------------------------------------------------------
// Tensor Data Mover: DMA one 64x64 bf16 tile (row pitch `pitch_elems` in
// memory) into LDS at byte offset lds_off, inserting 16B of LDS padding per
// 128B row (-> TPITCH).  D# built per CDNA5 ISA 8.3/8.4.  Tracked: TENSORcnt.
// ---------------------------------------------------------------------------
static __device__ __forceinline__ void tdm_load_tile64(unsigned lds_off,
                                                       const bf16_t* gsrc,
                                                       unsigned pitch_elems) {
  const unsigned long long ga = (unsigned long long)(uintptr_t)gsrc;
  v4u g0;
  g0[0] = 1u;                                   // count=1, user descriptor
  g0[1] = lds_off;                              // lds_addr (bytes)
  g0[2] = (unsigned)(ga & 0xFFFFFFFFu);         // global_addr[31:0]
  g0[3] = (unsigned)((ga >> 32) & 0x01FFFFFFu)  // global_addr[56:32]
          | (2u << 30);                         // type = 2 ("image")
  v8i g1;
  g1[0] = (int)((1u << 16)      // data_size = 1 -> 2 bytes
                | (1u << 20)    // pad_enable
                | (4u << 22)    // pad_interval code 4 -> every 32 DWORDs (128B)
                | (3u << 25));  // pad_amount  code 3 -> 4 DWORDs (16B)
  g1[1] = (int)((pitch_elems & 0xFFFFu) << 16);            // tensor_dim0 lo
  g1[2] = (int)((pitch_elems >> 16) & 0xFFFFu)             // tensor_dim0 hi
          | (int)(64u << 16);                              // tensor_dim1 lo = 64
  g1[3] = (int)(64u << 16);                                // dim1 hi=0, tile_dim0=64
  g1[4] = 64;                                              // tile_dim1=64, tile_dim2=0
  g1[5] = (int)pitch_elems;                                // tensor_dim0_stride lo
  g1[6] = 0;                                               // stride hi / dim1_stride lo
  g1[7] = 0;
  const v4i z4 = {0, 0, 0, 0};
  const v8i z8 = {0, 0, 0, 0, 0, 0, 0, 0};
  __builtin_amdgcn_tensor_load_to_lds(g0, g1, z4, z4, z8, 0);
}

static __device__ __forceinline__ unsigned lds_byte_off(const void* p) {
  return (unsigned)(uintptr_t)p;  // low 32 bits of generic addr = LDS offset
}

// ---------------------------------------------------------------------------
__global__ void f32_to_bf16(const float* __restrict__ in, bf16_t* __restrict__ out,
                            int n) {
  for (int i = blockIdx.x * blockDim.x + threadIdx.x; i < n;
       i += gridDim.x * blockDim.x)
    out[i] = (bf16_t)in[i];
}

// ---------------------------------------------------------------------------
// Shared GEMM core: C(64x64) += A(64xK) * B(64xK)^T with TDM-staged, double-
// buffered LDS tiles.  Each of the 8 waves owns one 16-row strip (msub) and a
// 32-col strip (npair) of the output.
// ---------------------------------------------------------------------------
struct GemmAcc { v8f a0, a1; };

static __device__ __forceinline__ GemmAcc gemm64x64(
    const bf16_t* __restrict__ Aglob, const bf16_t* __restrict__ Bglob,
    int K, bf16_t* Atile, bf16_t* Btile,   // [2][64*TPITCH] each
    int lane, int wave, int msub, int npair) {
  GemmAcc acc; acc.a0 = (v8f){}; acc.a1 = (v8f){};
  const int nsteps = K / 64;

  if (wave == 0) {
    tdm_load_tile64(lds_byte_off(Atile), Aglob, (unsigned)K);
    tdm_load_tile64(lds_byte_off(Btile), Bglob, (unsigned)K);
  }
  for (int kt = 0; kt < nsteps; ++kt) {
    const int cur = kt & 1;
    if (wave == 0) {
      if (kt + 1 < nsteps) {
        const int nxt = 1 - cur;
        tdm_load_tile64(lds_byte_off(Atile + nxt * 64 * TPITCH),
                        Aglob + (kt + 1) * 64, (unsigned)K);
        tdm_load_tile64(lds_byte_off(Btile + nxt * 64 * TPITCH),
                        Bglob + (kt + 1) * 64, (unsigned)K);
        __builtin_amdgcn_s_wait_tensorcnt(2);  // current buffer's 2 DMAs done
      } else {
        __builtin_amdgcn_s_wait_tensorcnt(0);
      }
    }
    __syncthreads();  // tiles visible to all waves
    const bf16_t* As = Atile + cur * 64 * TPITCH;
    const bf16_t* Bs = Btile + cur * 64 * TPITCH;
#pragma unroll
    for (int ks = 0; ks < 64; ks += 32) {
      v16bf a  = load_frag_a(As, TPITCH, msub * 16, ks, lane);
      v16bf b0 = load_frag_b(Bs, TPITCH, npair * 32,      ks, lane);
      v16bf b1 = load_frag_b(Bs, TPITCH, npair * 32 + 16, ks, lane);
      acc.a0 = wmma_bf(a, b0, acc.a0);
      acc.a1 = wmma_bf(a, b1, acc.a1);
    }
    __syncthreads();  // all reads done before buffer is overwritten
  }
  return acc;
}

// ---------------------------------------------------------------------------
// QKV GEMM: qkv[m,n] = sum_k x[m,k]*w_qkv[n,k] + b[n]  (NT, both K-major).
// N-tile = 64 = Hd, so each block owns one complete head slice -> fused
// per-head LayerNorm on the fp32 accumulators for the q/k sections.
// Outputs: q_ln, k_ln in [B,H,S,Hd] bf16; v transposed to [B,H,Hd,S] bf16.
// ---------------------------------------------------------------------------
__global__ __launch_bounds__(256) void qkv_gemm_ln(
    const bf16_t* __restrict__ xb, const bf16_t* __restrict__ wb,
    const float* __restrict__ bias,
    const float* __restrict__ qg, const float* __restrict__ qbeta,
    const float* __restrict__ kg, const float* __restrict__ kbeta,
    bf16_t* __restrict__ qout, bf16_t* __restrict__ kout,
    bf16_t* __restrict__ vT) {
  __shared__ bf16_t Atile[2 * 64 * TPITCH];
  __shared__ bf16_t Btile[2 * 64 * TPITCH];
  __shared__ float  tile[64][68];

  const int lane  = threadIdx.x & 31;
  const int wave  = threadIdx.x >> 5;
  const int msub  = wave & 3;
  const int npair = wave >> 2;
  const int Nbase = blockIdx.x * 64;
  const int Mbase = blockIdx.y * 64;

  GemmAcc acc = gemm64x64(xb + (size_t)Mbase * DD, wb + (size_t)Nbase * DD, DD,
                          Atile, Btile, lane, wave, msub, npair);

  {  // accumulators -> LDS with bias (D layout: m = r+(L&16?8:0), n = L&15)
    const int lr_off = (lane & 16) ? 8 : 0;
    const int lc0 = npair * 32 + (lane & 15);
    const int lc1 = lc0 + 16;
#pragma unroll
    for (int r = 0; r < 8; ++r) {
      int lr = msub * 16 + r + lr_off;
      tile[lr][lc0] = acc.a0[r] + bias[Nbase + lc0];
      tile[lr][lc1] = acc.a1[r] + bias[Nbase + lc1];
    }
  }
  __syncthreads();

  const int section = Nbase / DD;          // 0=q 1=k 2=v
  const int h       = (Nbase % DD) / HD;
  const int row     = threadIdx.x >> 2;    // token within tile
  const int quad    = threadIdx.x & 3;     // 4 threads per row
  const int token   = Mbase + row;
  const int bidx    = token >> 11;
  const int s       = token & (SS - 1);
  const size_t headTok = (size_t)(bidx * HH + h) * SS + s;

  if (section < 2) {
    const float* g  = (section == 0) ? qg : kg;
    const float* be = (section == 0) ? qbeta : kbeta;
    bf16_t* dst     = (section == 0) ? qout : kout;
    float sm = 0.f, sq = 0.f;
#pragma unroll
    for (int c = 0; c < 16; ++c) {
      float v = tile[row][quad * 16 + c];
      sm += v; sq += v * v;
    }
    sm += __shfl_xor(sm, 1, 32); sm += __shfl_xor(sm, 2, 32);
    sq += __shfl_xor(sq, 1, 32); sq += __shfl_xor(sq, 2, 32);
    const float mean = sm * (1.0f / 64.0f);
    const float var  = sq * (1.0f / 64.0f) - mean * mean;
    const float rstd = rsqrtf(var + 1e-5f);
#pragma unroll
    for (int c = 0; c < 16; ++c) {
      int d = quad * 16 + c;
      float v = (tile[row][d] - mean) * rstd * g[d] + be[d];
      dst[headTok * HD + d] = (bf16_t)v;
    }
  } else {  // v: store transposed [B,H,Hd,S] so PV B-fragments are contiguous
#pragma unroll
    for (int c = 0; c < 16; ++c) {
      int d = quad * 16 + c;
      vT[((size_t)(bidx * HH + h) * HD + d) * SS + s] = (bf16_t)tile[row][d];
    }
  }
}

// ---------------------------------------------------------------------------
// Attention per (b,h): 32 query rows per block, exact softmax.  The whole
// 32 x 2048 fp32 score strip lives in LDS (~257 KB of the WGP's 320 KB).
// ---------------------------------------------------------------------------
#define SLD (SS + 4)   // padded row stride (floats)

__global__ __launch_bounds__(256) void attention(
    const bf16_t* __restrict__ qb, const bf16_t* __restrict__ kb,
    const bf16_t* __restrict__ vT, bf16_t* __restrict__ aout) {
  extern __shared__ float Sb[];          // [32][SLD]
  __shared__ float rsum[32];

  const int lane  = threadIdx.x & 31;
  const int wave  = threadIdx.x >> 5;
  const int bh    = blockIdx.y;
  const int Qbase = blockIdx.x * 32;
  const bf16_t* qh = qb + (size_t)bh * SS * HD;
  const bf16_t* kh = kb + (size_t)bh * SS * HD;
  const bf16_t* vh = vT + (size_t)bh * HD * SS;
  const int lr_off = (lane & 16) ? 8 : 0;

  // ---- Phase 1: scores.  Wave w owns key columns [w*256, w*256+256).
  v16bf qf[2][2];
#pragma unroll
  for (int mg = 0; mg < 2; ++mg)
#pragma unroll
    for (int ds = 0; ds < 2; ++ds)
      qf[mg][ds] = load_frag_a(qh, HD, Qbase + mg * 16, ds * 32, lane);

  const int kcol0 = wave * 256;
  for (int t = 0; t < 16; ++t) {
    const int jb = kcol0 + t * 16;
    v16bf b0 = load_frag_b(kh, HD, jb, 0, lane);   // B[d][j] from k[j][d]
    v16bf b1 = load_frag_b(kh, HD, jb, 32, lane);
#pragma unroll
    for (int mg = 0; mg < 2; ++mg) {
      v8f acc = {};
      acc = wmma_bf(qf[mg][0], b0, acc);
      acc = wmma_bf(qf[mg][1], b1, acc);
      const int col = jb + (lane & 15);
#pragma unroll
      for (int r = 0; r < 8; ++r)
        Sb[(size_t)(mg * 16 + r + lr_off) * SLD + col] = acc[r] * 0.125f;
    }
  }
  __syncthreads();

  // ---- Phase 2: softmax.  8 threads per row, 256 columns each.
  {
    const int row = threadIdx.x >> 3;
    const int sub = threadIdx.x & 7;
    float* Sr = Sb + (size_t)row * SLD + sub * 256;
    float mx = -3.0e38f;
    for (int c = 0; c < 256; ++c) mx = fmaxf(mx, Sr[c]);
    mx = fmaxf(mx, __shfl_xor(mx, 1, 32));
    mx = fmaxf(mx, __shfl_xor(mx, 2, 32));
    mx = fmaxf(mx, __shfl_xor(mx, 4, 32));
    float sm = 0.f;
    for (int c = 0; c < 256; ++c) {
      float e = __expf(Sr[c] - mx);
      Sr[c] = e;
      sm += e;
    }
    sm += __shfl_xor(sm, 1, 32);
    sm += __shfl_xor(sm, 2, 32);
    sm += __shfl_xor(sm, 4, 32);
    if (sub == 0) rsum[row] = sm;
  }
  __syncthreads();

  // ---- Phase 3: O = P V.  Each wave owns one 16x16 output tile, K=2048.
  {
    const int mg = wave & 1;
    const int dsub = wave >> 1;
    v8f acc = {};
    for (int jb = 0; jb < SS; jb += 32) {
      v16bf a = load_frag_a_f32(Sb, SLD, mg * 16, jb, lane);
      v16bf b = load_frag_b(vh, SS, dsub * 16, jb, lane);  // B[j][d] from vT[d][j]
      acc = wmma_bf(a, b, acc);
    }
    const int d  = dsub * 16 + (lane & 15);
    const int b_ = bh >> 4, h = bh & 15;
#pragma unroll
    for (int r = 0; r < 8; ++r) {
      const int lr = mg * 16 + r + lr_off;
      const int srow = Qbase + lr;
      const float v = acc[r] / rsum[lr];
      aout[((size_t)(b_ * SS + srow)) * DD + h * HD + d] = (bf16_t)v;
    }
  }
}

// ---------------------------------------------------------------------------
// Output projection: out[m,n] = sum_k attn[m,k]*w_proj[n,k] + b[n] (fp32 out)
// ---------------------------------------------------------------------------
__global__ __launch_bounds__(256) void proj_gemm(
    const bf16_t* __restrict__ ab, const bf16_t* __restrict__ wb,
    const float* __restrict__ bias, float* __restrict__ out) {
  __shared__ bf16_t Atile[2 * 64 * TPITCH];
  __shared__ bf16_t Btile[2 * 64 * TPITCH];

  const int lane  = threadIdx.x & 31;
  const int wave  = threadIdx.x >> 5;
  const int msub  = wave & 3;
  const int npair = wave >> 2;
  const int Nbase = blockIdx.x * 64;
  const int Mbase = blockIdx.y * 64;

  GemmAcc acc = gemm64x64(ab + (size_t)Mbase * DD, wb + (size_t)Nbase * DD, DD,
                          Atile, Btile, lane, wave, msub, npair);

  const int lr_off = (lane & 16) ? 8 : 0;
  const int c0 = Nbase + npair * 32 + (lane & 15);
  const int c1 = c0 + 16;
#pragma unroll
  for (int r = 0; r < 8; ++r) {
    const int row = Mbase + msub * 16 + r + lr_off;
    out[(size_t)row * DD + c0] = acc.a0[r] + bias[c0];
    out[(size_t)row * DD + c1] = acc.a1[r] + bias[c1];
  }
}

// ---------------------------------------------------------------------------
extern "C" void kernel_launch(void* const* d_in, const int* in_sizes, int n_in,
                              void* d_out, int out_size, void* d_ws, size_t ws_size,
                              hipStream_t stream) {
  (void)in_sizes; (void)n_in; (void)out_size; (void)ws_size;
  const float* x      = (const float*)d_in[0];
  const float* w_qkv  = (const float*)d_in[1];
  const float* b_qkv  = (const float*)d_in[2];
  const float* w_proj = (const float*)d_in[3];
  const float* b_proj = (const float*)d_in[4];
  const float* qg     = (const float*)d_in[5];
  const float* qbeta  = (const float*)d_in[6];
  const float* kg     = (const float*)d_in[7];
  const float* kbeta  = (const float*)d_in[8];
  float* out = (float*)d_out;

  char* ws = (char*)d_ws;
  bf16_t* xb     = (bf16_t*)ws; ws += (size_t)TOK * DD * sizeof(bf16_t);
  bf16_t* wqkvb  = (bf16_t*)ws; ws += (size_t)N3 * DD * sizeof(bf16_t);
  bf16_t* wprojb = (bf16_t*)ws; ws += (size_t)DD * DD * sizeof(bf16_t);
  bf16_t* qbuf   = (bf16_t*)ws; ws += (size_t)TOK * DD * sizeof(bf16_t);
  bf16_t* kbuf   = (bf16_t*)ws; ws += (size_t)TOK * DD * sizeof(bf16_t);
  bf16_t* vTbuf  = (bf16_t*)ws; ws += (size_t)TOK * DD * sizeof(bf16_t);
  bf16_t* abuf   = (bf16_t*)ws; ws += (size_t)TOK * DD * sizeof(bf16_t);

  f32_to_bf16<<<1024, 256, 0, stream>>>(x, xb, TOK * DD);
  f32_to_bf16<<<1024, 256, 0, stream>>>(w_qkv, wqkvb, N3 * DD);
  f32_to_bf16<<<1024, 256, 0, stream>>>(w_proj, wprojb, DD * DD);

  qkv_gemm_ln<<<dim3(N3 / 64, TOK / 64), 256, 0, stream>>>(
      xb, wqkvb, b_qkv, qg, qbeta, kg, kbeta, qbuf, kbuf, vTbuf);

  attention<<<dim3(SS / 32, BB * HH), 256, (size_t)32 * SLD * sizeof(float),
              stream>>>(qbuf, kbuf, vTbuf, abuf);

  proj_gemm<<<dim3(DD / 64, TOK / 64), 256, 0, stream>>>(abuf, wprojb, b_proj, out);
}
